// Int4Linear_69956427317863
// MI455X (gfx1250) — matrix operand order repaired and checked by reading, not compiled, not run on hardware
//
#include <hip/hip_runtime.h>

#define OUT_F 16384
#define IN_F  4096
#define GS    128
#define NG    32      // IN_F / GS
#define BTOK  32      // token batch

typedef __attribute__((ext_vector_type(16))) _Float16 v16h;
typedef __attribute__((ext_vector_type(8)))  _Float16 v8h;
typedef __attribute__((ext_vector_type(2)))  _Float16 h2;
typedef __attribute__((ext_vector_type(8)))  float    v8f;
typedef __attribute__((ext_vector_type(4)))  unsigned v4u;

union FragB { v16h v; h2 p[8]; };
union FragA { v16h v; v8h h[2]; };
union U32H2 { unsigned u; h2 h; };

// Kernel 1: convert x f32 -> f16 into workspace, applying the K-permutation
// (within each 4-element chunk: 0,1,2,3 -> 0,2,1,3) that matches the byte
// order produced by the packed int4 dequant trick in kernel 2. Permuting K
// identically in A and B leaves A x B unchanged.
__global__ __launch_bounds__(256) void Int4Linear_xcvt_kernel(
    const float* __restrict__ x, _Float16* __restrict__ xf) {
  int tid = blockIdx.x * 256 + threadIdx.x;   // 0 .. 32*4096-1
  int k   = tid & (IN_F - 1);
  int j   = k & 3;
  int src = (k & ~3) | ((j & 1) << 1) | (j >> 1);  // swap 1<->2
  xf[tid] = (_Float16)x[(tid - k) + src];
}

// (n ^ 8) | 0x6400 == f16(1032 + v) exactly for the signed nibble v;
// subtract 1032 (exact in f16), then scale. Lowers to v_bitop3_b32 +
// v_pk_add_f16 + v_pk_mul_f16 per f16 pair.
__device__ __forceinline__ FragB dequant16(v4u q, h2 s2, h2 c1032) {
  FragB b;
#pragma unroll
  for (int d = 0; d < 4; ++d) {
    unsigned qq = q[d];
    U32H2 e, o;
    e.u = ( qq       & 0x000F000Fu) ^ 0x64086408u;  // bytes 0,2
    o.u = ((qq >> 8) & 0x000F000Fu) ^ 0x64086408u;  // bytes 1,3
    b.p[2 * d]     = (e.h - c1032) * s2;
    b.p[2 * d + 1] = (o.h - c1032) * s2;
  }
  return b;
}

// Kernel 2: weight-streaming int4 GEMM via v_wmma_f32_16x16x32_f16.
// One wave: 16 output columns x all 32 rows (two 16x16 C tiles).
// Block = 8 waves = 128 columns; grid = 16384/128 = 128 blocks.
// Two K=32 steps are processed per unrolled iteration with BOTH B fragments
// live across the 4 interleaved WMMAs, forcing the register allocator to use
// distinct octets and letting real work cover the WMMA->VALU WAR hazard.
__global__ __launch_bounds__(256) void Int4Linear_gemm_kernel(
    const _Float16* __restrict__ xf,      // [32][4096] f16, K-permuted
    const signed char* __restrict__ w,    // [16384][4096] int8 (int4 values)
    const float* __restrict__ scales,     // [16384][32]
    const float* __restrict__ bias,       // [16384]
    float* __restrict__ out) {            // [32][16384]
  const int lane = threadIdx.x & 31;
  const int wave = threadIdx.x >> 5;
  const int half = lane >> 4;     // 0: lanes 0-15, 1: lanes 16-31
  const int l16  = lane & 15;
  const int col  = blockIdx.x * 128 + wave * 16 + l16;

  // B fragment source: lanes 0-15 hold K=kk+0..15, lanes 16-31 K=kk+16..31
  const signed char* wrow = w + (size_t)col * IN_F + half * 16;
  // A fragment source: lo lanes K=kk+0..15, hi lanes K=kk+16..31
  const _Float16* xlo = xf + (size_t)l16 * IN_F + half * 16;  // rows M=0..15
  const _Float16* xhi = xlo + (size_t)16 * IN_F;              // rows M=16..31

  v8f acc0 = {};   // M = 0..15 tile
  v8f acc1 = {};   // M = 16..31 tile
  const h2 c1032 = { (_Float16)1032.0f, (_Float16)1032.0f };

  for (int g = 0; g < NG; ++g) {
    const float s = scales[col * NG + g];
    const _Float16 sh = (_Float16)s;
    const h2 s2 = { sh, sh };

    // stream-prefetch the weight row two scale-groups ahead
    const int pg = (g + 2 < NG) ? (g + 2) : (NG - 1);
    __builtin_prefetch(wrow + pg * GS, 0, 0);

    // load the whole group's weights for this lane (4 x B128)
    const signed char* wg = wrow + g * GS;
    v4u q[4];
    q[0] = *(const v4u*)(wg);
    q[1] = *(const v4u*)(wg + 32);
    q[2] = *(const v4u*)(wg + 64);
    q[3] = *(const v4u*)(wg + 96);

#pragma unroll
    for (int tt = 0; tt < 4; tt += 2) {
      const int kkA = g * GS + tt * 32;
      const int kkB = kkA + 32;

      // Both B fragments live at once -> forced distinct register octets.
      FragB bA = dequant16(q[tt],     s2, c1032);
      FragB bB = dequant16(q[tt + 1], s2, c1032);

      // A fragments for both K-steps (two B128 loads per tile per step)
      FragA a0A, a1A, a0B, a1B;
      a0A.h[0] = *(const v8h*)(xlo + kkA);
      a0A.h[1] = *(const v8h*)(xlo + kkA + 8);
      a1A.h[0] = *(const v8h*)(xhi + kkA);
      a1A.h[1] = *(const v8h*)(xhi + kkA + 8);
      a0B.h[0] = *(const v8h*)(xlo + kkB);
      a0B.h[1] = *(const v8h*)(xlo + kkB + 8);
      a1B.h[0] = *(const v8h*)(xhi + kkB);
      a1B.h[1] = *(const v8h*)(xhi + kkB + 8);

      acc0 = __builtin_amdgcn_wmma_f32_16x16x32_f16(
          false, a0A.v, false, bA.v, (short)0, acc0, false, false);
      acc1 = __builtin_amdgcn_wmma_f32_16x16x32_f16(
          false, a1A.v, false, bA.v, (short)0, acc1, false, false);
      acc0 = __builtin_amdgcn_wmma_f32_16x16x32_f16(
          false, a0B.v, false, bB.v, (short)0, acc0, false, false);
      acc1 = __builtin_amdgcn_wmma_f32_16x16x32_f16(
          false, a1B.v, false, bB.v, (short)0, acc1, false, false);
    }
  }

  // C/D layout: VGPR r -> M = r + 8*half (lanes 0-15) / +8 (lanes 16-31)
  const float bv = bias[col];
#pragma unroll
  for (int r = 0; r < 8; ++r) {
    const int m0 = r + 8 * half;
    out[(size_t)m0 * OUT_F + col]        = acc0[r] + bv;
    out[(size_t)(m0 + 16) * OUT_F + col] = acc1[r] + bv;
  }
}

extern "C" void kernel_launch(void* const* d_in, const int* in_sizes, int n_in,
                              void* d_out, int out_size, void* d_ws, size_t ws_size,
                              hipStream_t stream) {
  const float*       x      = (const float*)d_in[0];
  const signed char* w      = (const signed char*)d_in[1];
  const float*       scales = (const float*)d_in[2];
  const float*       bias   = (const float*)d_in[3];
  float*             out    = (float*)d_out;
  _Float16*          xf     = (_Float16*)d_ws;   // 32*4096*2 = 256 KB

  Int4Linear_xcvt_kernel<<<(BTOK * IN_F) / 256, 256, 0, stream>>>(x, xf);
  Int4Linear_gemm_kernel<<<OUT_F / 128, 256, 0, stream>>>(xf, w, scales, bias, out);
}
